// TransformerEncoder_3762391352013
// MI455X (gfx1250) — matrix operand order repaired and checked by reading, not compiled, not run on hardware
//
#include <hip/hip_runtime.h>
#include <hip/hip_bf16.h>

typedef unsigned short u16;
typedef unsigned int   u32;
typedef __attribute__((ext_vector_type(16))) __bf16 v16bf;
typedef __attribute__((ext_vector_type(8)))  float  v8f;
typedef __attribute__((ext_vector_type(4)))  u32    u32x4;
typedef int v4i_g __attribute__((vector_size(16)));   // matches async builtin param

#define TT   768
#define DD   1024
#define HH   16
#define HDD  64
#define LL   14
#define FFD  4096
#define PP   1535          // 2T-1 rel-pos rows
#define PLD  1536          // padded ld for bd buffer

// CDNA5 async global->LDS path (ASYNCcnt), guarded so compile never regresses.
#if defined(__AMDGCN__) && defined(__gfx1250__) && \
    __has_builtin(__builtin_amdgcn_global_load_async_to_lds_b128) && \
    __has_builtin(__builtin_amdgcn_s_wait_asynccnt)
#define USE_ASYNC 1
#else
#define USE_ASYNC 0
#endif

#define AS1 __attribute__((address_space(1)))
#define AS3 __attribute__((address_space(3)))

__device__ __forceinline__ u16 f2bf(float f) {
  u32 u = __float_as_uint(f);
  u32 r = u + 0x7FFFu + ((u >> 16) & 1u);   // round-to-nearest-even
  return (u16)(r >> 16);
}

union FragAB { v16bf v; u32x4 q[2]; };

// ---------------------------------------------------------------------------
// Generic batched bf16 WMMA GEMM:  C = A(MxK) * B + bias + resid  [relu]
//   bTrans=0: B is KxN row-major (weights)   bTrans=1: B is NxK row-major (k/p)
//   C written as f32 (Cf) or bf16 (Cb). Per-batch (head) strides sA/sB/sC.
//   Interior NN tiles use double-buffered GLOBAL_LOAD_ASYNC_TO_LDS_B128.
// ---------------------------------------------------------------------------
#define BM 128
#define BN 128
#define BK 32

// fragments per ISA 7.12.2: A lanes0-15 K0-7/16-23, lanes16-31 K8-15/24-31;
// B lane%16==N col, K in same order as A (lanes0-15 K0-7/16-23, lanes16-31 K8-15/24-31).
__device__ __forceinline__ void wmma_tile(
    const u16* sA, const u16* sB, int lane, int wm, int wn, v8f (&acc)[2][4])
{
  FragAB aF[2], bF[4];
#pragma unroll
  for (int mi = 0; mi < 2; ++mi) {
    int row  = wm + mi * 16 + (lane & 15);
    int half = (lane >> 4) * 8;
    aF[mi].q[0] = *(const u32x4*)(sA + row * BK + half);
    aF[mi].q[1] = *(const u32x4*)(sA + row * BK + 16 + half);
  }
#pragma unroll
  for (int ni = 0; ni < 4; ++ni) {
    int col = wn + ni * 16 + (lane & 15);
    int half = (lane >> 4) * 8;
#pragma unroll
    for (int s = 0; s < 16; ++s)
      ((u16*)&bF[ni])[s] = sB[(half + (s & 7) + (s >> 3) * 16) * BN + col];
  }
#pragma unroll
  for (int mi = 0; mi < 2; ++mi)
#pragma unroll
    for (int ni = 0; ni < 4; ++ni)
      acc[mi][ni] = __builtin_amdgcn_wmma_f32_16x16x32_bf16(
          false, aF[mi].v, false, bF[ni].v, (short)0, acc[mi][ni], false, false);
}

__global__ __launch_bounds__(256) void gemm_bf16_wmma(
    const u16* __restrict__ A, const u16* __restrict__ B,
    float* Cf, u16* Cb,
    const float* __restrict__ bias, const float* __restrict__ resid,
    int M, int N, int K, int lda, int ldb, int ldc,
    long long sA, long long sB, long long sC,
    int bTrans, int doRelu)
{
  __shared__ __align__(16) u16 lA[2][BM * BK];
  __shared__ __align__(16) u16 lB[2][BK * BN];

  const int z = blockIdx.z;
  A += (long long)z * sA;
  B += (long long)z * sB;
  const long long coff = (long long)z * sC;

  const int m0   = blockIdx.y * BM;
  const int n0   = blockIdx.x * BN;
  const int tid  = threadIdx.x;
  const int lane = tid & 31;
  const int wid  = tid >> 5;
  const int wm   = (wid & 3) * 32;   // 4 wave-rows of 32
  const int wn   = (wid >> 2) * 64;  // 2 wave-cols of 64

  v8f acc[2][4] = {};
  const int nk = K / BK;

#if USE_ASYNC
  const bool asyncOK = (!bTrans) && (m0 + BM <= M) && (n0 + BN <= N);
  if (asyncOK) {
    // Per-wave: exactly 4 async b128 ops per stage (2 for A, 2 for B).
    auto stage_async = [&](int k0, int buf) {
#pragma unroll
      for (int i = 0; i < 2; ++i) {
        int c = tid + i * 256, r = c >> 2, col = (c & 3) * 8;
        __builtin_amdgcn_global_load_async_to_lds_b128(
            (AS1 v4i_g*)(A + (long long)(m0 + r) * lda + k0 + col),
            (AS3 v4i_g*)(&lA[buf][r * BK + col]), 0, 0);
      }
#pragma unroll
      for (int i = 0; i < 2; ++i) {
        int c = tid + i * 256, kk = c >> 4, col = (c & 15) * 8;
        __builtin_amdgcn_global_load_async_to_lds_b128(
            (AS1 v4i_g*)(B + (long long)(k0 + kk) * ldb + n0 + col),
            (AS3 v4i_g*)(&lB[buf][kk * BN + col]), 0, 0);
      }
    };
    stage_async(0, 0);
    for (int i = 0; i < nk; ++i) {
      const int cur = i & 1;
      __syncthreads();                       // all reads of buf cur^1 finished
      if (i + 1 < nk) {
        stage_async((i + 1) * BK, cur ^ 1);  // prefetch next tile
        __builtin_amdgcn_s_wait_asynccnt(4); // in-order: first 4 (cur) done
      } else {
        __builtin_amdgcn_s_wait_asynccnt(0);
      }
      __syncthreads();                       // cur staged by all waves
      wmma_tile(lA[cur], lB[cur], lane, wm, wn, acc);
    }
  } else
#endif
  {
    // synchronous fallback: edge tiles and NT (transposed-B) attention GEMMs
    for (int k0 = 0; k0 < K; k0 += BK) {
      __syncthreads();
#pragma unroll
      for (int i = 0; i < 2; ++i) {
        int c   = tid + i * 256;             // 512 chunks of 8 bf16
        int r   = c >> 2;
        int col = (c & 3) * 8;
        u32x4 val = {0u, 0u, 0u, 0u};
        int gm = m0 + r;
        if (gm < M) val = *(const u32x4*)(A + (long long)gm * lda + k0 + col);
        *(u32x4*)(&lA[0][r * BK + col]) = val;
      }
      if (!bTrans) {
#pragma unroll
        for (int i = 0; i < 2; ++i) {
          int c   = tid + i * 256;
          int kk  = c >> 4;
          int col = (c & 15) * 8;
          u32x4 val = {0u, 0u, 0u, 0u};
          int gn = n0 + col;
          if (gn < N) val = *(const u32x4*)(B + (long long)(k0 + kk) * ldb + gn);
          *(u32x4*)(&lB[0][kk * BN + col]) = val;
        }
      } else {
        // B memory is N x K; transpose during staging: lB[k][n] = B[n][k]
        int n  = tid >> 1;
        int kc = (tid & 1) * 16;
        int gn = n0 + n;
        u16 tmp[16];
        if (gn < N) {
          *(u32x4*)(tmp)     = *(const u32x4*)(B + (long long)gn * ldb + k0 + kc);
          *(u32x4*)(tmp + 8) = *(const u32x4*)(B + (long long)gn * ldb + k0 + kc + 8);
        } else {
#pragma unroll
          for (int j = 0; j < 16; ++j) tmp[j] = 0;
        }
#pragma unroll
        for (int j = 0; j < 16; ++j) lB[0][(kc + j) * BN + n] = tmp[j];
      }
      __syncthreads();
      wmma_tile(lA[0], lB[0], lane, wm, wn, acc);
    }
  }

  // ---- epilogue: C VGPRj -> lanes0-15 M=j, lanes16-31 M=8+j ----
#pragma unroll
  for (int mi = 0; mi < 2; ++mi) {
#pragma unroll
    for (int ni = 0; ni < 4; ++ni) {
#pragma unroll
      for (int j = 0; j < 8; ++j) {
        int gm = m0 + wm + mi * 16 + ((lane >> 4) << 3) + j;
        int gn = n0 + wn + ni * 16 + (lane & 15);
        if (gm < M && gn < N) {
          float v = acc[mi][ni][j];
          if (bias)   v += bias[gn];
          if (resid)  v += resid[(long long)gm * ldc + gn];
          if (doRelu) v = fmaxf(v, 0.0f);
          long long o = coff + (long long)gm * ldc + gn;
          if (Cf) Cf[o] = v;
          else    Cb[o] = f2bf(v);
        }
      }
    }
  }
}

// ---------------------------------------------------------------------------
// Row LayerNorm over D: out = (relu?)(LN(x)*g+b) * scale ; f32 and/or bf16 out
// ---------------------------------------------------------------------------
__global__ __launch_bounds__(256) void layernorm_k(
    const float* __restrict__ x, const float* __restrict__ g,
    const float* __restrict__ b, float* outF, u16* outB,
    int D, float eps, int doRelu, float scale)
{
  const int row = blockIdx.x;
  const float* xr = x + (long long)row * D;
  __shared__ float red[256];
  const int tid = threadIdx.x;
  float s = 0.f;
  for (int c = tid; c < D; c += 256) s += xr[c];
  red[tid] = s; __syncthreads();
  for (int o = 128; o > 0; o >>= 1) { if (tid < o) red[tid] += red[tid + o]; __syncthreads(); }
  float mean = red[0] / D;
  __syncthreads();
  float vs = 0.f;
  for (int c = tid; c < D; c += 256) { float d = xr[c] - mean; vs += d * d; }
  red[tid] = vs; __syncthreads();
  for (int o = 128; o > 0; o >>= 1) { if (tid < o) red[tid] += red[tid + o]; __syncthreads(); }
  float inv = rsqrtf(red[0] / D + eps);
  for (int c = tid; c < D; c += 256) {
    float v = (xr[c] - mean) * inv * g[c] + b[c];
    if (doRelu) v = fmaxf(v, 0.f);
    v *= scale;
    long long o = (long long)row * D + c;
    if (outF) outF[o] = v;
    if (outB) outB[o] = f2bf(v);
  }
}

// ---------------------------------------------------------------------------
// Fused rel-shift + combine + softmax:  probs[h,t,s] = softmax_s((ac+bd_shift)*scale)
// bd_shift index: n = s + (T-1) - t  (Transformer-XL rel_shift)
// ---------------------------------------------------------------------------
__global__ __launch_bounds__(256) void softmax_shift(
    const float* __restrict__ ac, const float* __restrict__ bd,
    u16* __restrict__ probs, int T, int S, int bdld, float scale)
{
  const int t = blockIdx.x, h = blockIdx.y;
  const long long roA = ((long long)h * T + t) * S;
  const long long roB = ((long long)h * T + t) * bdld + (T - 1 - t);
  __shared__ float red[256];
  const int tid = threadIdx.x;
  float v[3];
  float m = -1e30f;
#pragma unroll
  for (int i = 0; i < 3; ++i) {
    int s = tid + i * 256;
    float x = (s < S) ? (ac[roA + s] + bd[roB + s]) * scale : -1e30f;
    v[i] = x; m = fmaxf(m, x);
  }
  red[tid] = m; __syncthreads();
  for (int o = 128; o > 0; o >>= 1) { if (tid < o) red[tid] = fmaxf(red[tid], red[tid + o]); __syncthreads(); }
  m = red[0]; __syncthreads();
  float sum = 0.f;
#pragma unroll
  for (int i = 0; i < 3; ++i) { v[i] = __expf(v[i] - m); sum += v[i]; }
  red[tid] = sum; __syncthreads();
  for (int o = 128; o > 0; o >>= 1) { if (tid < o) red[tid] += red[tid + o]; __syncthreads(); }
  float inv = 1.f / red[0];
#pragma unroll
  for (int i = 0; i < 3; ++i) {
    int s = tid + i * 256;
    if (s < S) probs[roA + s] = f2bf(v[i] * inv);
  }
}

// ---------------------------------------------------------------------------
// Espnet rel positional encoding -> bf16 (2T-1, D)
// row r<T: pe_pos[T-1-r]; r>=T: pe_neg[r-T+1] (sin gets the sign)
// ---------------------------------------------------------------------------
__global__ __launch_bounds__(256) void posenc_k(u16* __restrict__ pos, int T, int D)
{
  long long idx = (long long)blockIdx.x * 256 + threadIdx.x;
  long long total = (long long)(2 * T - 1) * D;
  if (idx >= total) return;
  int r = (int)(idx / D), c = (int)(idx % D);
  int i2 = c & ~1;
  float div = __expf(-(float)i2 * (9.210340371976184f / (float)D));  // ln(1e4)
  float p, sign;
  if (r < T) { p = (float)(T - 1 - r); sign = 1.f; }
  else       { p = (float)(r - T + 1); sign = -1.f; }
  float ang = p * div;
  float v = (c & 1) ? __cosf(ang) : sign * __sinf(ang);
  pos[idx] = f2bf(v);
}

__global__ __launch_bounds__(256) void cvt_bf16(
    const float* __restrict__ in, u16* __restrict__ out, long long n)
{
  long long i = (long long)blockIdx.x * 256 + threadIdx.x;
  if (i < n) out[i] = f2bf(in[i]);
}

__global__ __launch_bounds__(256) void addbias_cvt(
    const float* __restrict__ in, const float* __restrict__ bias,
    u16* __restrict__ out, long long n, int mask)
{
  long long i = (long long)blockIdx.x * 256 + threadIdx.x;
  if (i < n) out[i] = f2bf(in[i] + bias[i & mask]);
}

// ---------------------------------------------------------------------------
// Host-side helpers
// ---------------------------------------------------------------------------
static inline void run_gemm(hipStream_t s, const u16* A, const u16* B,
    float* Cf, u16* Cb, const float* bias, const float* resid,
    int M, int N, int K, int lda, int ldb, int ldc,
    long long sA, long long sB, long long sC, int bTrans, int relu, int batch)
{
  dim3 g((N + BN - 1) / BN, (M + BM - 1) / BM, batch);
  gemm_bf16_wmma<<<g, dim3(256), 0, s>>>(A, B, Cf, Cb, bias, resid,
      M, N, K, lda, ldb, ldc, sA, sB, sC, bTrans, relu);
}

static inline void run_cvt(hipStream_t s, const float* in, u16* out, long long n)
{
  cvt_bf16<<<dim3((unsigned)((n + 255) / 256)), dim3(256), 0, s>>>(in, out, n);
}

extern "C" void kernel_launch(void* const* d_in, const int* in_sizes, int n_in,
                              void* d_out, int out_size, void* d_ws, size_t ws_size,
                              hipStream_t stream)
{
  (void)in_sizes; (void)n_in; (void)out_size; (void)ws_size;
  const float* xs         = (const float*)d_in[0];
  const float* embed_W    = (const float*)d_in[1];
  const float* embed_b    = (const float*)d_in[2];
  const float* embed_ln_g = (const float*)d_in[3];
  const float* embed_ln_b = (const float*)d_in[4];
  const float* Wq   = (const float*)d_in[5];
  const float* bq   = (const float*)d_in[6];
  const float* Wk   = (const float*)d_in[7];
  const float* bk   = (const float*)d_in[8];
  const float* Wv   = (const float*)d_in[9];
  const float* bv   = (const float*)d_in[10];
  const float* Wo   = (const float*)d_in[11];
  const float* bo   = (const float*)d_in[12];
  const float* Wp   = (const float*)d_in[13];
  const float* pos_u = (const float*)d_in[14];
  const float* pos_v = (const float*)d_in[15];
  const float* ln1_g = (const float*)d_in[16];
  const float* ln1_b = (const float*)d_in[17];
  const float* ln2_g = (const float*)d_in[18];
  const float* ln2_b = (const float*)d_in[19];
  const float* ffW1  = (const float*)d_in[20];
  const float* ffb1  = (const float*)d_in[21];
  const float* ffW2  = (const float*)d_in[22];
  const float* ffb2  = (const float*)d_in[23];
  const float* after_g = (const float*)d_in[24];
  const float* after_b = (const float*)d_in[25];

  // ---- workspace bump allocator (~166 MB total) ----
  char* wsp = (char*)d_ws;
  size_t off = 0;
  auto alloc = [&](size_t bytes) -> void* {
    void* p = wsp + off; off = (off + bytes + 255) & ~(size_t)255; return p;
  };
  float* x    = (float*)alloc((size_t)TT * DD * 4);          // residual stream
  float* tmpf = (float*)alloc((size_t)TT * DD * 4);          // embed pre-LN / q f32
  u16* xnb    = (u16*)alloc((size_t)TT * DD * 2);            // LN out (bf16 GEMM A)
  u16* kb     = (u16*)alloc((size_t)TT * DD * 2);
  u16* vb     = (u16*)alloc((size_t)TT * DD * 2);
  u16* qub    = (u16*)alloc((size_t)TT * DD * 2);            // q + pos_u
  u16* qvb    = (u16*)alloc((size_t)TT * DD * 2);            // q + pos_v
  u16* posb   = (u16*)alloc((size_t)PP * DD * 2);            // rel-pos table bf16
  u16* pb     = (u16*)alloc((size_t)PP * DD * 2);            // pos @ Wp
  float* ac   = (float*)alloc((size_t)HH * TT * TT * 4);     // content scores
  float* bdf  = (float*)alloc((size_t)HH * TT * PLD * 4);    // unshifted pos scores
  u16* probs  = (u16*)alloc((size_t)HH * TT * TT * 2);       // softmax probs bf16
  u16* ob     = (u16*)alloc((size_t)TT * DD * 2);            // attn output bf16
  u16* ffhb   = (u16*)alloc((size_t)TT * FFD * 2);           // relu(ffn1) bf16
  u16* wbuf   = (u16*)alloc((size_t)DD * FFD * 2);           // weight bf16 staging

  const long long nTD = (long long)TT * DD;

  // positional encoding table (once)
  {
    long long n = (long long)PP * DD;
    posenc_k<<<dim3((unsigned)((n + 255) / 256)), dim3(256), 0, stream>>>(posb, TT, DD);
  }

  // ---- embed: relu(LN(xs@W+b)) * sqrt(D) ----
  run_cvt(stream, xs, xnb, nTD);
  run_cvt(stream, embed_W, wbuf, (long long)DD * DD);
  run_gemm(stream, xnb, wbuf, tmpf, nullptr, embed_b, nullptr,
           TT, DD, DD, DD, DD, DD, 0, 0, 0, 0, 0, 1);
  layernorm_k<<<dim3(TT), dim3(256), 0, stream>>>(tmpf, embed_ln_g, embed_ln_b,
           x, (u16*)nullptr, DD, 1e-5f, 1, 32.0f);   // sqrt(1024)=32

  const float scale = 0.125f;  // HD^-0.5
  for (int i = 0; i < LL; ++i) {
    const long long wOff  = (long long)i * DD * DD;
    // LN1 -> bf16
    layernorm_k<<<dim3(TT), dim3(256), 0, stream>>>(x, ln1_g + i * DD, ln1_b + i * DD,
        (float*)nullptr, xnb, DD, 1e-12f, 0, 1.0f);
    // q (f32), k,v (bf16)
    run_cvt(stream, Wq + wOff, wbuf, (long long)DD * DD);
    run_gemm(stream, xnb, wbuf, tmpf, nullptr, bq + i * DD, nullptr,
             TT, DD, DD, DD, DD, DD, 0, 0, 0, 0, 0, 1);
    run_cvt(stream, Wk + wOff, wbuf, (long long)DD * DD);
    run_gemm(stream, xnb, wbuf, nullptr, kb, bk + i * DD, nullptr,
             TT, DD, DD, DD, DD, DD, 0, 0, 0, 0, 0, 1);
    run_cvt(stream, Wv + wOff, wbuf, (long long)DD * DD);
    run_gemm(stream, xnb, wbuf, nullptr, vb, bv + i * DD, nullptr,
             TT, DD, DD, DD, DD, DD, 0, 0, 0, 0, 0, 1);
    // p = pos @ Wp  (M=1535, guarded)
    run_cvt(stream, Wp + wOff, wbuf, (long long)DD * DD);
    run_gemm(stream, posb, wbuf, nullptr, pb, nullptr, nullptr,
             PP, DD, DD, DD, DD, DD, 0, 0, 0, 0, 0, 1);
    // qu = q + pos_u[i], qv = q + pos_v[i]  (broadcast over t)
    {
      unsigned blocks = (unsigned)((nTD + 255) / 256);
      addbias_cvt<<<dim3(blocks), dim3(256), 0, stream>>>(tmpf, pos_u + i * DD, qub, nTD, DD - 1);
      addbias_cvt<<<dim3(blocks), dim3(256), 0, stream>>>(tmpf, pos_v + i * DD, qvb, nTD, DD - 1);
    }
    // ac[h] = qu_h @ k_h^T   (NT, batched over heads)
    run_gemm(stream, qub, kb, ac, nullptr, nullptr, nullptr,
             TT, TT, HDD, DD, DD, TT, HDD, HDD, (long long)TT * TT, 1, 0, HH);
    // bd[h] = qv_h @ p_h^T   (NT, N=2T-1, padded ldc)
    run_gemm(stream, qvb, pb, bdf, nullptr, nullptr, nullptr,
             TT, PP, HDD, DD, DD, PLD, HDD, HDD, (long long)TT * PLD, 1, 0, HH);
    // softmax((ac + rel_shift(bd)) * scale) -> bf16 probs
    softmax_shift<<<dim3(TT, HH), dim3(256), 0, stream>>>(ac, bdf, probs, TT, TT, PLD, scale);
    // o_h = probs_h @ v_h    (NN, N=64, batched)
    run_gemm(stream, probs, vb, nullptr, ob, nullptr, nullptr,
             TT, HDD, TT, TT, DD, DD, (long long)TT * TT, HDD, HDD, 0, 0, HH);
    // x = x + o @ Wo + bo
    run_cvt(stream, Wo + wOff, wbuf, (long long)DD * DD);
    run_gemm(stream, ob, wbuf, x, nullptr, bo + i * DD, x,
             TT, DD, DD, DD, DD, DD, 0, 0, 0, 0, 0, 1);
    // FFN
    layernorm_k<<<dim3(TT), dim3(256), 0, stream>>>(x, ln2_g + i * DD, ln2_b + i * DD,
        (float*)nullptr, xnb, DD, 1e-12f, 0, 1.0f);
    run_cvt(stream, ffW1 + (long long)i * DD * FFD, wbuf, (long long)DD * FFD);
    run_gemm(stream, xnb, wbuf, nullptr, ffhb, ffb1 + i * FFD, nullptr,
             TT, FFD, DD, DD, FFD, FFD, 0, 0, 0, 0, 1, 1);
    run_cvt(stream, ffW2 + (long long)i * FFD * DD, wbuf, (long long)FFD * DD);
    run_gemm(stream, ffhb, wbuf, x, nullptr, ffb2 + i * DD, x,
             TT, DD, FFD, FFD, DD, DD, 0, 0, 0, 0, 0, 1);
  }
  // final LN -> d_out (f32)
  layernorm_k<<<dim3(TT), dim3(256), 0, stream>>>(x, after_g, after_b,
      (float*)d_out, (u16*)nullptr, DD, 1e-5f, 0, 1.0f);
}
